// GraphAttention_63797444215664
// MI455X (gfx1250) — matrix operand order repaired and checked by reading, not compiled, not run on hardware
//
#include <hip/hip_runtime.h>
#include <hip/hip_bf16.h>

typedef __attribute__((ext_vector_type(16))) _Float16 v16h;
typedef __attribute__((ext_vector_type(8)))  _Float16 v8h;
typedef __attribute__((ext_vector_type(8)))  float    v8f;

#define NN      4096
#define NFEAT   512
#define NHID    128
#define NCLASS  40
#define NCLASSP 48
#define NHEADS  4

// ---------------------------------------------------------------------------
// Elementwise f32 -> f16 convert
// ---------------------------------------------------------------------------
__global__ void gat_cvt_f16_kernel(const float* __restrict__ in,
                                   _Float16* __restrict__ out, int n) {
  int i = blockIdx.x * 256 + threadIdx.x;
  if (i < n) out[i] = (_Float16)in[i];
}

// ---------------------------------------------------------------------------
// Transpose + convert: in f32 [batch][R][C] -> out f16 [batch][Cp][R], zero pad
// ---------------------------------------------------------------------------
__global__ void gat_transpose_cvt_kernel(const float* __restrict__ in,
                                         _Float16* __restrict__ out,
                                         int R, int C, int Cp,
                                         long strideIn, long strideOut) {
  int r = blockIdx.x * 16 + threadIdx.x;
  int c = blockIdx.y * 16 + threadIdx.y;
  const float* inb = in + (long)blockIdx.z * strideIn;
  _Float16* outb = out + (long)blockIdx.z * strideOut;
  if (r < R && c < Cp) {
    float v = (c < C) ? inb[(long)r * C + c] : 0.0f;
    outb[(long)c * R + r] = (_Float16)v;
  }
}

// ---------------------------------------------------------------------------
// WMMA GEMM: C[b][M][N] (f32) = A[b][M][K] (f16, row-major) * Bt[b][N][K]^T
// One wave per 16 x (NTW*16) C strip; A fragment reused across NTW n-tiles.
// A fragment (16-bit 16x32): lane<16 -> M=lane, K in {0..7,16..23};
//                            lane>=16 -> M=lane-16, K in {8..15,24..31}.
// B fragment: lane<16 -> N=lane, K in {0..7,16..23}; lane>=16 -> N=lane-16, K in {8..15,24..31}.
// ---------------------------------------------------------------------------
template <int NTW>
__global__ void gat_wmma_gemm_kernel(const _Float16* __restrict__ A,
                                     const _Float16* __restrict__ Bt,
                                     float* __restrict__ C,
                                     int M, int N, int K,
                                     long strideA, long strideB, long strideC) {
  const int lane = threadIdx.x;
  const int half = lane >> 4;
  const int lid  = lane & 15;
  const int tm = blockIdx.x, tn0 = blockIdx.y * NTW, b = blockIdx.z;

  const _Float16* Ab  = A  + (long)b * strideA + (long)(tm * 16 + lid) * K;
  const _Float16* Bb0 = Bt + (long)b * strideB + (long)(tn0 * 16 + lid) * K;
  float* Cb = C + (long)b * strideC;

  const int abase = half * 8;    // A k-offset within 32-step
  const int bbase = half * 8;    // B k-offset within 32-step

  v8f acc[NTW];
#pragma unroll
  for (int t = 0; t < NTW; ++t) acc[t] = (v8f){};

  for (int kt = 0; kt < K; kt += 32) {
    v8h a0 = *(const v8h*)(Ab + kt + abase);
    v8h a1 = *(const v8h*)(Ab + kt + abase + 16);
    v16h av;
#pragma unroll
    for (int e = 0; e < 8; ++e) { av[e] = a0[e]; av[e + 8] = a1[e]; }
#pragma unroll
    for (int t = 0; t < NTW; ++t) {
      const _Float16* Bb = Bb0 + (long)t * 16 * K;
      v8h b0 = *(const v8h*)(Bb + kt + bbase);
      v8h b1 = *(const v8h*)(Bb + kt + bbase + 16);
      v16h bv;
#pragma unroll
      for (int e = 0; e < 8; ++e) { bv[e] = b0[e]; bv[e + 8] = b1[e]; }
      acc[t] = __builtin_amdgcn_wmma_f32_16x16x32_f16(false, av, false, bv,
                                                      (short)0, acc[t], false, false);
    }
  }
  // C/D layout: VGPR r -> M = r + 8*half, N = lid
#pragma unroll
  for (int t = 0; t < NTW; ++t)
#pragma unroll
    for (int r = 0; r < 8; ++r)
      Cb[(long)(tm * 16 + r + half * 8) * N + (tn0 + t) * 16 + lid] = acc[t][r];
}

// ---------------------------------------------------------------------------
// src/dst GEMVs: src[b][n] = h[b][n][:F].a1 ; dst = .a2  (a2 = avec+F)
// ---------------------------------------------------------------------------
__global__ void gat_dot_kernel(const float* __restrict__ h,
                               const float* __restrict__ avec,
                               float* __restrict__ src, float* __restrict__ dst,
                               int F, int rowstride) {
  int n = blockIdx.x, b = blockIdx.y, t = threadIdx.x;
  const float* hp = h + ((long)b * NN + n) * rowstride;
  const float* a1 = avec + (long)b * 2 * F;
  const float* a2 = a1 + F;
  float p1 = 0.f, p2 = 0.f;
  for (int f = t; f < F; f += 64) {
    float v = hp[f];
    p1 += v * a1[f];
    p2 += v * a2[f];
  }
  __shared__ float s1[64], s2[64];
  s1[t] = p1; s2[t] = p2;
  __syncthreads();
  for (int off = 32; off > 0; off >>= 1) {
    if (t < off) { s1[t] += s1[t + off]; s2[t] += s2[t + off]; }
    __syncthreads();
  }
  if (t == 0) { src[(long)b * NN + n] = s1[0]; dst[(long)b * NN + n] = s2[0]; }
}

// ---------------------------------------------------------------------------
// Pass 1: per-row online softmax stats (max, sum) for NH heads in one adj
// sweep. 4 consecutive j per thread -> int4/float4 vector loads.
// ---------------------------------------------------------------------------
template <int NH>
__global__ void gat_stats_kernel(const int* __restrict__ adj,
                                 const float* __restrict__ src,
                                 const float* __restrict__ dst,
                                 float* __restrict__ rmax,
                                 float* __restrict__ rsum) {
  const int i = blockIdx.x, tid = threadIdx.x;
  float sl[NH], m[NH], s[NH];
#pragma unroll
  for (int h = 0; h < NH; ++h) { sl[h] = src[h * NN + i]; m[h] = -3.0e38f; s[h] = 0.f; }
  const int* arow = adj + (long)i * NN;
  for (int j = tid * 4; j < NN; j += 1024) {
    if (j + 1024 < NN) __builtin_prefetch(arow + j + 1024, 0, 1);
    int4 q = *(const int4*)(arow + j);
    int aa[4] = {q.x, q.y, q.z, q.w};
#pragma unroll
    for (int h = 0; h < NH; ++h) {
      float4 d = *(const float4*)(dst + (long)h * NN + j);
      float dd[4] = {d.x, d.y, d.z, d.w};
#pragma unroll
      for (int u = 0; u < 4; ++u) {
        float z = sl[h] + dd[u];
        z = z > 0.f ? z : 0.1f * z;
        float e = (aa[u] > 0) ? z : -9.0e15f;
        if (e > m[h]) { s[h] = s[h] * __expf(m[h] - e) + 1.f; m[h] = e; }
        else          { s[h] += __expf(e - m[h]); }
      }
    }
  }
  __shared__ float smx[NH * 256], ssm[NH * 256];
#pragma unroll
  for (int h = 0; h < NH; ++h) { smx[h * 256 + tid] = m[h]; ssm[h * 256 + tid] = s[h]; }
  __syncthreads();
  for (int off = 128; off > 0; off >>= 1) {
    if (tid < off) {
#pragma unroll
      for (int h = 0; h < NH; ++h) {
        float m1 = smx[h * 256 + tid],       s1 = ssm[h * 256 + tid];
        float m2 = smx[h * 256 + tid + off], s2 = ssm[h * 256 + tid + off];
        float M = fmaxf(m1, m2);
        smx[h * 256 + tid] = M;
        ssm[h * 256 + tid] = s1 * __expf(m1 - M) + s2 * __expf(m2 - M);
      }
    }
    __syncthreads();
  }
  if (tid == 0) {
#pragma unroll
    for (int h = 0; h < NH; ++h) {
      rmax[h * NN + i] = smx[h * 256];
      rsum[h * NN + i] = ssm[h * 256];
    }
  }
}

// ---------------------------------------------------------------------------
// Pass 2 (fused): O[b][i][f] = softmax_row(P) @ H  without materializing P.
// 8 waves split K=4096; each wave builds the exp() A-fragment in registers
// and runs NT WMMA n-tiles; cross-wave tree reduction in LDS; 1/s in epilogue.
// ---------------------------------------------------------------------------
template <int NT>
__global__ void gat_aggregate_kernel(const int* __restrict__ adj,
                                     const float* __restrict__ src,
                                     const float* __restrict__ dst,
                                     const float* __restrict__ rmax,
                                     const float* __restrict__ rsum,
                                     const _Float16* __restrict__ Ht, // [b][NT*16][NN]
                                     float* __restrict__ O) {         // [b][NN][NT*16]
  const int lane = threadIdx.x & 31;
  const int wave = threadIdx.x >> 5;  // 0..7
  const int half = lane >> 4;
  const int lid  = lane & 15;
  const int rowt = blockIdx.x;
  const int b    = blockIdx.y;

  const int i = rowt * 16 + lid;             // A-fragment row owned by this lane
  const float si = src[(long)b * NN + i];
  const float mi = rmax[(long)b * NN + i];
  const float* dstb = dst + (long)b * NN;
  const int* adjrow = adj + (long)i * NN;
  const _Float16* Htb = Ht + (long)b * NT * 16 * NN;

  const int j0 = wave * (NN / 8);
  const int abase = half * 8;
  const int bbase = half * 8;

  v8f acc[NT];
#pragma unroll
  for (int t = 0; t < NT; ++t) acc[t] = (v8f){};

  for (int jt = 0; jt < NN / 8; jt += 32) {
    const int ja = j0 + jt + abase;  // A k-runs: [ja..ja+7], [ja+16..ja+23]
    int4   q0 = *(const int4*)(adjrow + ja);
    int4   q1 = *(const int4*)(adjrow + ja + 4);
    int4   q2 = *(const int4*)(adjrow + ja + 16);
    int4   q3 = *(const int4*)(adjrow + ja + 20);
    float4 d0 = *(const float4*)(dstb + ja);
    float4 d1 = *(const float4*)(dstb + ja + 4);
    float4 d2 = *(const float4*)(dstb + ja + 16);
    float4 d3 = *(const float4*)(dstb + ja + 20);
    int   am[16] = {q0.x, q0.y, q0.z, q0.w, q1.x, q1.y, q1.z, q1.w,
                    q2.x, q2.y, q2.z, q2.w, q3.x, q3.y, q3.z, q3.w};
    float dv[16] = {d0.x, d0.y, d0.z, d0.w, d1.x, d1.y, d1.z, d1.w,
                    d2.x, d2.y, d2.z, d2.w, d3.x, d3.y, d3.z, d3.w};
    v16h av;
#pragma unroll
    for (int e = 0; e < 16; ++e) {
      float z = si + dv[e];
      z = z > 0.f ? z : 0.1f * z;
      float ev = (am[e] > 0) ? z : -9.0e15f;
      av[e] = (_Float16)__expf(ev - mi);  // unnormalized softmax weight
    }
    const int jb = j0 + jt + bbase;
#pragma unroll
    for (int t = 0; t < NT; ++t) {
      const _Float16* hp = Htb + (long)(t * 16 + lid) * NN + jb;
      v8h b0 = *(const v8h*)(hp);
      v8h b1 = *(const v8h*)(hp + 16);
      v16h bv;
#pragma unroll
      for (int e = 0; e < 8; ++e) { bv[e] = b0[e]; bv[e + 8] = b1[e]; }
      acc[t] = __builtin_amdgcn_wmma_f32_16x16x32_f16(false, av, false, bv,
                                                      (short)0, acc[t], false, false);
    }
  }

  // cross-wave tree reduction: 8 -> 4 -> 2 -> 1
  __shared__ float sred[4][16][NT * 16];
  for (int hstep = 4; hstep >= 1; hstep >>= 1) {
    if (wave >= hstep && wave < 2 * hstep) {
#pragma unroll
      for (int t = 0; t < NT; ++t)
#pragma unroll
        for (int r = 0; r < 8; ++r)
          sred[wave - hstep][r + half * 8][t * 16 + lid] = acc[t][r];
    }
    __syncthreads();
    if (wave < hstep) {
#pragma unroll
      for (int t = 0; t < NT; ++t)
#pragma unroll
        for (int r = 0; r < 8; ++r)
          acc[t][r] += sred[wave][r + half * 8][t * 16 + lid];
    }
    __syncthreads();
  }

  if (wave == 0) {
#pragma unroll
    for (int t = 0; t < NT; ++t)
#pragma unroll
      for (int r = 0; r < 8; ++r) {
        int m = rowt * 16 + r + half * 8;
        float inv = 1.0f / rsum[(long)b * NN + m];
        O[((long)b * NN + m) * (NT * 16) + t * 16 + lid] = acc[t][r] * inv;
      }
  }
}

// ---------------------------------------------------------------------------
// Double ELU + head concat: O1 [4][NN][128] f32 -> xcat [NN][512] f16
// ---------------------------------------------------------------------------
__global__ void gat_eluconcat_kernel(const float* __restrict__ O1,
                                     _Float16* __restrict__ xcat) {
  int idx = blockIdx.x * 256 + threadIdx.x;  // over NN*512
  int nrow = idx >> 9, col = idx & 511;
  int hh = col >> 7, f = col & 127;
  float v = O1[((long)hh * NN + nrow) * NHID + f];
  v = v > 0.f ? v : __expf(v) - 1.f;
  v = v > 0.f ? v : __expf(v) - 1.f;
  xcat[idx] = (_Float16)v;
}

// ---------------------------------------------------------------------------
// Final ELU, drop padding: O2 [NN][48] -> out [NN][40]
// ---------------------------------------------------------------------------
__global__ void gat_elu_out_kernel(const float* __restrict__ O2,
                                   float* __restrict__ out, int total) {
  int idx = blockIdx.x * 256 + threadIdx.x;
  if (idx >= total) return;
  int nrow = idx / NCLASS, c = idx % NCLASS;
  float v = O2[(long)nrow * NCLASSP + c];
  out[idx] = v > 0.f ? v : __expf(v) - 1.f;
}

// ---------------------------------------------------------------------------
extern "C" void kernel_launch(void* const* d_in, const int* in_sizes, int n_in,
                              void* d_out, int out_size, void* d_ws, size_t ws_size,
                              hipStream_t stream) {
  const float* x   = (const float*)d_in[0];   // [4096][512]
  const int*   adj = (const int*)d_in[1];     // [4096][4096]
  const float* W0  = (const float*)d_in[2];   // [4][512][128]
  const float* a0  = (const float*)d_in[3];   // [4][256][1]
  const float* Wo  = (const float*)d_in[4];   // [512][40]
  const float* ao  = (const float*)d_in[5];   // [80][1]
  float* out = (float*)d_out;                 // [4096][40]

  char* base = (char*)d_ws;
  size_t off = 0;
  auto take = [&](size_t bytes) -> char* {
    off = (off + 255) & ~(size_t)255;
    char* p = base + off;
    off += bytes;
    return p;
  };
  _Float16* xh    = (_Float16*)take((size_t)NN * NFEAT * 2);
  _Float16* W0t   = (_Float16*)take((size_t)NHEADS * NHID * NFEAT * 2);
  float*    hbuf  = (float*)   take((size_t)NHEADS * NN * NHID * 4);
  _Float16* hht   = (_Float16*)take((size_t)NHEADS * NHID * NN * 2);
  float*    src0  = (float*)   take((size_t)NHEADS * NN * 4);
  float*    dst0  = (float*)   take((size_t)NHEADS * NN * 4);
  float*    m0    = (float*)   take((size_t)NHEADS * NN * 4);
  float*    s0    = (float*)   take((size_t)NHEADS * NN * 4);
  float*    O1    = (float*)   take((size_t)NHEADS * NN * NHID * 4);
  _Float16* xcat  = (_Float16*)take((size_t)NN * NFEAT * 2);
  _Float16* Wot   = (_Float16*)take((size_t)NCLASSP * NFEAT * 2);
  float*    h2    = (float*)   take((size_t)NN * NCLASSP * 4);
  _Float16* hh2t  = (_Float16*)take((size_t)NCLASSP * NN * 2);
  float*    src2  = (float*)   take((size_t)NN * 4);
  float*    dst2  = (float*)   take((size_t)NN * 4);
  float*    m2    = (float*)   take((size_t)NN * 4);
  float*    s2    = (float*)   take((size_t)NN * 4);
  float*    O2    = (float*)   take((size_t)NN * NCLASSP * 4);
  (void)ws_size; (void)n_in; (void)in_sizes; (void)out_size;

  // ---- Layer 1 ----
  gat_cvt_f16_kernel<<<(NN * NFEAT) / 256, 256, 0, stream>>>(x, xh, NN * NFEAT);

  gat_transpose_cvt_kernel<<<dim3(NFEAT / 16, NHID / 16, NHEADS), dim3(16, 16), 0, stream>>>(
      W0, W0t, NFEAT, NHID, NHID, (long)NFEAT * NHID, (long)NHID * NFEAT);

  // h[b] = x @ W0[b]; each wave computes a 16x64 strip (4 n-tiles)
  gat_wmma_gemm_kernel<4><<<dim3(NN / 16, NHID / 64, NHEADS), 32, 0, stream>>>(
      xh, W0t, hbuf, NN, NHID, NFEAT,
      /*strideA=*/0, (long)NHID * NFEAT, (long)NN * NHID);

  gat_transpose_cvt_kernel<<<dim3(NN / 16, NHID / 16, NHEADS), dim3(16, 16), 0, stream>>>(
      hbuf, hht, NN, NHID, NHID, (long)NN * NHID, (long)NHID * NN);

  gat_dot_kernel<<<dim3(NN, NHEADS), 64, 0, stream>>>(hbuf, a0, src0, dst0, NHID, NHID);

  gat_stats_kernel<NHEADS><<<NN, 256, 0, stream>>>(adj, src0, dst0, m0, s0);

  gat_aggregate_kernel<NHID / 16><<<dim3(NN / 16, NHEADS), 256, 0, stream>>>(
      adj, src0, dst0, m0, s0, hht, O1);

  gat_eluconcat_kernel<<<(NN * NFEAT) / 256, 256, 0, stream>>>(O1, xcat);

  // ---- Layer 2 ----
  gat_transpose_cvt_kernel<<<dim3(NFEAT / 16, NCLASSP / 16, 1), dim3(16, 16), 0, stream>>>(
      Wo, Wot, NFEAT, NCLASS, NCLASSP, 0, 0);

  // h2 = xcat @ Wo; one wave per 16-row strip covering all 48 padded cols
  gat_wmma_gemm_kernel<3><<<dim3(NN / 16, 1, 1), 32, 0, stream>>>(
      xcat, Wot, h2, NN, NCLASSP, NFEAT, 0, 0, 0);

  gat_transpose_cvt_kernel<<<dim3(NN / 16, NCLASSP / 16, 1), dim3(16, 16), 0, stream>>>(
      h2, hh2t, NN, NCLASSP, NCLASSP, 0, 0);

  gat_dot_kernel<<<dim3(NN, 1), 64, 0, stream>>>(h2, ao, src2, dst2, NCLASS, NCLASSP);

  gat_stats_kernel<1><<<NN, 256, 0, stream>>>(adj, src2, dst2, m2, s2);

  gat_aggregate_kernel<NCLASSP / 16><<<dim3(NN / 16, 1), 256, 0, stream>>>(
      adj, src2, dst2, m2, s2, hh2t, O2);

  gat_elu_out_kernel<<<(NN * NCLASS + 255) / 256, 256, 0, stream>>>(O2, out, NN * NCLASS);
}
